// TransformerModel_11716670783707
// MI455X (gfx1250) — matrix operand order repaired and checked by reading, not compiled, not run on hardware
//
#include <hip/hip_runtime.h>

typedef __bf16 bf16_t;
typedef __attribute__((ext_vector_type(16))) __bf16 bf16x16;
typedef __attribute__((ext_vector_type(8)))  __bf16 bf16x8;
typedef __attribute__((ext_vector_type(4)))  __bf16 bf16x4;
typedef __attribute__((ext_vector_type(8)))  float  f32x8;
typedef __attribute__((ext_vector_type(4)))  float  f32x4;
typedef __attribute__((ext_vector_type(4)))  unsigned int u32x4;
typedef __attribute__((ext_vector_type(8)))  int    i32x8;
typedef __attribute__((ext_vector_type(4)))  int    i32x4;

#define EMBED 1024
#define NB 4
#define SEQ 4096
#define NTOK (NB * SEQ)
#define VOCAB 32000
#define KSPLIT 16

#if defined(__has_builtin)
#if __has_builtin(__builtin_amdgcn_tensor_load_to_lds) && \
    __has_builtin(__builtin_amdgcn_s_wait_tensorcnt)
#define USE_TDM 1
#endif
#endif
#ifndef USE_TDM
#define USE_TDM 0
#endif

// ---------------------------------------------------------------------------
// 1) h = embed[x] : gather to bf16 (GEMM A operand); fp32 last-token rows kept
//    for the residual path.
// ---------------------------------------------------------------------------
__global__ __launch_bounds__(256) void gather_embed_kernel(
    const int* __restrict__ x, const float* __restrict__ embedW,
    bf16_t* __restrict__ hbf, float* __restrict__ hlast)
{
    size_t t  = (size_t)blockIdx.x * blockDim.x + threadIdx.x; // NTOK*256 threads
    int d4    = (int)(t & 255);
    int tok   = (int)(t >> 8);
    int idx   = x[tok];
    f32x4 v   = *(const f32x4*)(embedW + (size_t)idx * EMBED + d4 * 4);
    bf16x4 o;
    o[0] = (bf16_t)v[0]; o[1] = (bf16_t)v[1]; o[2] = (bf16_t)v[2]; o[3] = (bf16_t)v[3];
    *(bf16x4*)(hbf + (size_t)tok * EMBED + d4 * 4) = o;
    if ((tok & (SEQ - 1)) == SEQ - 1) {
        int b = tok >> 12;
        *(f32x4*)(hlast + b * EMBED + d4 * 4) = v;
    }
}

// ---------------------------------------------------------------------------
// 2) Weight transpose: src (K x N fp32, row-major) -> dst (N x K bf16).
// ---------------------------------------------------------------------------
__global__ __launch_bounds__(256) void transpose_bf16_kernel(
    const float* __restrict__ src, bf16_t* __restrict__ dst, int Kdim, int Ndim)
{
    size_t t = (size_t)blockIdx.x * blockDim.x + threadIdx.x; // (Kdim/4)*Ndim threads
    int n  = (int)(t % (size_t)Ndim);
    int k4 = (int)(t / (size_t)Ndim);
    bf16x4 o;
#pragma unroll
    for (int i = 0; i < 4; ++i)
        o[i] = (bf16_t)src[(size_t)(k4 * 4 + i) * Ndim + n];
    *(bf16x4*)(dst + (size_t)n * Kdim + k4 * 4) = o;
}

// ---------------------------------------------------------------------------
// WMMA fragment helper (this file's A/B k order: one contiguous 16-elem K run/lane)
// ---------------------------------------------------------------------------
__device__ inline bf16x16 load_a_frag(const bf16_t* base)
{
    bf16x8 lo = *(const bf16x8*)base;
    bf16x8 hi = *(const bf16x8*)(base + 8);
    return __builtin_shufflevector(lo, hi, 0, 1, 2, 3, 4, 5, 6, 7,
                                   8, 9, 10, 11, 12, 13, 14, 15);
}

// ---------------------------------------------------------------------------
// 3) KV projection GEMM: (16384 x 1024) @ (1024 x 2048) bf16 WMMA, fp32 acc.
//    Wave tile 64x64 (4x4 frags), block = 8 waves -> 256(M) x 128(N).
//    B tile (128 x 32 bf16 = 8 KB) staged into LDS by the Tensor Data Mover
//    each K-step; all 8 waves read fragments from LDS.
// ---------------------------------------------------------------------------
__global__ __launch_bounds__(256) void gemm_kv_wmma_kernel(
    const bf16_t* __restrict__ A,    // NTOK x EMBED
    const bf16_t* __restrict__ Bt,   // 2048 x EMBED (k,v weight rows, transposed)
    const float*  __restrict__ bias, // 2048 (qkv_b + 1024)
    bf16_t* __restrict__ Kout,       // NTOK x EMBED
    bf16_t* __restrict__ Vout)       // NTOK x EMBED
{
    const int Kd = EMBED;
    int lane = threadIdx.x & 31;
    int wid  = threadIdx.x >> 5;
    int wm = wid & 3, wn = wid >> 2;
    int m0 = blockIdx.x * 256 + wm * 64;
    int nb = blockIdx.y * 128;           // block N base
    int n0 = nb + wn * 64;               // wave N base
    int l15 = lane & 15, half = lane >> 4;

    __shared__ bf16_t Bsh[128 * 32];     // 128 N-rows x 32 K, row stride 32

    f32x8 acc[4][4] = {};

#if USE_TDM
    // Tensor DMA descriptor, 2D tile: X = 32 K elems (64B), Y = 128 N rows.
    unsigned int lds_base = (unsigned int)(uintptr_t)&Bsh[0];
    // group1: data_size=2B(code 1); tensor_dim0=1024, tensor_dim1=2048;
    // tile_dim0=32, tile_dim1=128; tensor_dim0_stride=1024.
    i32x8 g1 = { (int)(1u << 16),
                 (int)((1024u & 0xFFFFu) << 16),
                 (int)((2048u & 0xFFFFu) << 16),
                 (int)(32u << 16),
                 (int)128,
                 (int)1024,
                 0, 0 };
    i32x4 g2 = { 0, 0, 0, 0 };
    i32x4 g3 = { 0, 0, 0, 0 };
#endif

    for (int k0 = 0; k0 < Kd; k0 += 32) {
        bf16x16 a[4], b[4];
#pragma unroll
        for (int mi = 0; mi < 4; ++mi)
            a[mi] = load_a_frag(A + (size_t)(m0 + mi * 16 + l15) * Kd + k0 + half * 16);
        if (k0 + 32 < Kd) // global_prefetch_b8 for next A K-slab
            __builtin_prefetch(A + (size_t)(m0 + l15) * Kd + k0 + 32, 0, 3);

#if USE_TDM
        __syncthreads();                 // previous tile fully consumed
        if (wid == 0) {
            unsigned long long ga =
                (unsigned long long)(uintptr_t)(Bt + (size_t)nb * Kd + k0);
            u32x4 g0 = { 1u,                                   // count=1
                         lds_base,                             // lds_addr
                         (unsigned int)ga,                     // global_addr lo
                         (unsigned int)((ga >> 32) & 0x01FFFFFFu) | (2u << 30) };
#if __clang_major__ >= 23
            i32x8 g4 = { 0, 0, 0, 0, 0, 0, 0, 0 };
            __builtin_amdgcn_tensor_load_to_lds(g0, g1, g2, g3, g4, 0);
#else
            __builtin_amdgcn_tensor_load_to_lds(g0, g1, g2, g3, 0);
#endif
            __builtin_amdgcn_s_wait_tensorcnt(0);
        }
        __syncthreads();                 // tile visible to all waves
#pragma unroll
        for (int ni = 0; ni < 4; ++ni)
            b[ni] = *(const bf16x16*)(Bsh + (wn * 64 + ni * 16 + l15) * 32 + half * 16);
#else
#pragma unroll
        for (int ni = 0; ni < 4; ++ni)
            b[ni] = *(const bf16x16*)(Bt + (size_t)(n0 + ni * 16 + l15) * Kd + k0 + half * 16);
#endif

#pragma unroll
        for (int mi = 0; mi < 4; ++mi)
#pragma unroll
            for (int ni = 0; ni < 4; ++ni)
                acc[mi][ni] = __builtin_amdgcn_wmma_f32_16x16x32_bf16(
                    false, a[mi], false, b[ni], (short)0, acc[mi][ni], false, false);
    }

#pragma unroll
    for (int ni = 0; ni < 4; ++ni) {
        int n = n0 + ni * 16 + l15;
        float bv = bias[n];
#pragma unroll
        for (int mi = 0; mi < 4; ++mi) {
#pragma unroll
            for (int r = 0; r < 8; ++r) {
                int m = m0 + mi * 16 + r + half * 8;
                bf16_t o = (bf16_t)(acc[mi][ni][r] + bv);
                if (n < EMBED) Kout[(size_t)m * EMBED + n] = o;
                else           Vout[(size_t)m * EMBED + (n - EMBED)] = o;
            }
        }
    }
}

// ---------------------------------------------------------------------------
// 4) q for the last token only: q[b,n] = hlast[b,:] . Wq_t[n,:] + bias
// ---------------------------------------------------------------------------
__global__ __launch_bounds__(256) void qlast_gemv_kernel(
    const float* __restrict__ hlast, const bf16_t* __restrict__ Wt,
    const float* __restrict__ qkv_b, float* __restrict__ qlast)
{
    int t = blockIdx.x * blockDim.x + threadIdx.x; // NB*EMBED
    int b = t >> 10, n = t & 1023;
    const float*  h = hlast + b * EMBED;
    const bf16_t* w = Wt + (size_t)n * EMBED;
    float acc = qkv_b[n];
    for (int d = 0; d < EMBED; d += 8) {
        bf16x8 wv = *(const bf16x8*)(w + d);
#pragma unroll
        for (int i = 0; i < 8; ++i) acc += h[d + i] * (float)wv[i];
    }
    qlast[t] = acc;
}

// ---------------------------------------------------------------------------
// 5) logits[b,k] = (q_last[b] . K[b,k]) / sqrt(1024)
// ---------------------------------------------------------------------------
__global__ __launch_bounds__(256) void logits_kernel(
    const float* __restrict__ qlast, const bf16_t* __restrict__ Kbf,
    float* __restrict__ logits)
{
    int t = blockIdx.x * blockDim.x + threadIdx.x; // NTOK
    int b = t >> 12;
    const float*  q  = qlast + b * EMBED;
    const bf16_t* kr = Kbf + (size_t)t * EMBED;
    float acc = 0.f;
    for (int d = 0; d < EMBED; d += 8) {
        bf16x8 kv = *(const bf16x8*)(kr + d);
#pragma unroll
        for (int i = 0; i < 8; ++i) acc += q[d + i] * (float)kv[i];
    }
    logits[t] = acc * 0.03125f; // 1/sqrt(1024)
}

// ---------------------------------------------------------------------------
// 6) softmax over 4096 keys, one block per batch
// ---------------------------------------------------------------------------
__global__ __launch_bounds__(256) void softmax_kernel(
    const float* __restrict__ logits, float* __restrict__ probs)
{
    __shared__ float red[256];
    int b = blockIdx.x, tid = threadIdx.x;
    const float* l = logits + b * SEQ;
    float* p = probs + b * SEQ;
    float mx = -3.4e38f;
    for (int i = tid; i < SEQ; i += 256) mx = fmaxf(mx, l[i]);
    red[tid] = mx; __syncthreads();
    for (int s = 128; s > 0; s >>= 1) {
        if (tid < s) red[tid] = fmaxf(red[tid], red[tid + s]);
        __syncthreads();
    }
    mx = red[0]; __syncthreads();
    float sum = 0.f;
    for (int i = tid; i < SEQ; i += 256) {
        float e = __expf(l[i] - mx);
        p[i] = e;
        sum += e;
    }
    red[tid] = sum; __syncthreads();
    for (int s = 128; s > 0; s >>= 1) {
        if (tid < s) red[tid] += red[tid + s];
        __syncthreads();
    }
    float inv = 1.f / red[0];
    for (int i = tid; i < SEQ; i += 256) p[i] *= inv;
}

// ---------------------------------------------------------------------------
// 7a) attn partial: part[kz][b][d] = sum over a 256-key chunk of p[k]*V[b,k,d]
//     256 blocks in flight -> enough concurrency to stream V at HBM rate.
// ---------------------------------------------------------------------------
__global__ __launch_bounds__(256) void attnout_partial_kernel(
    const float* __restrict__ probs, const bf16_t* __restrict__ Vbf,
    float* __restrict__ part)
{
    int b  = blockIdx.x;
    int d  = blockIdx.y * 256 + threadIdx.x;
    int kz = blockIdx.z;
    const float*  p = probs + b * SEQ + kz * (SEQ / KSPLIT);
    const bf16_t* v = Vbf + ((size_t)b * SEQ + (size_t)kz * (SEQ / KSPLIT)) * EMBED + d;
    float acc = 0.f;
    for (int k = 0; k < SEQ / KSPLIT; ++k)
        acc += p[k] * (float)v[(size_t)k * EMBED];
    part[((size_t)kz * NB + b) * EMBED + d] = acc;
}

// 7b) deterministic reduce over KSPLIT partials + residual, store bf16
__global__ __launch_bounds__(256) void attnout_reduce_kernel(
    const float* __restrict__ part, const float* __restrict__ hlast,
    bf16_t* __restrict__ alast)
{
    int t = blockIdx.x * blockDim.x + threadIdx.x; // NB*EMBED
    float acc = hlast[t];
#pragma unroll
    for (int kz = 0; kz < KSPLIT; ++kz)
        acc += part[(size_t)kz * (NB * EMBED) + t];
    alast[t] = (bf16_t)acc;
}

// ---------------------------------------------------------------------------
// 8) Final vocab GEMM: (4 x 1024) @ (1024 x 32000) with WMMA (A rows 4..15
//    zero-padded). One wave per 16x64 output tile.
// ---------------------------------------------------------------------------
__global__ __launch_bounds__(256) void final_gemm_wmma_kernel(
    const bf16_t* __restrict__ Albf, const bf16_t* __restrict__ Owt,
    const float* __restrict__ out_b, float* __restrict__ out)
{
    int lane = threadIdx.x & 31;
    int wid  = threadIdx.x >> 5;
    int n0 = (blockIdx.x * 8 + wid) * 64;
    if (n0 >= VOCAB) return; // wave-uniform
    int l15 = lane & 15, half = lane >> 4;
    f32x8 acc[4] = {};
    for (int k0 = 0; k0 < EMBED; k0 += 32) {
        bf16x16 a = {};
        if (l15 < NB)
            a = load_a_frag(Albf + (size_t)l15 * EMBED + k0 + half * 16);
#pragma unroll
        for (int ni = 0; ni < 4; ++ni) {
            bf16x16 bf = *(const bf16x16*)(Owt + (size_t)(n0 + ni * 16 + l15) * EMBED + k0 + half * 16);
            acc[ni] = __builtin_amdgcn_wmma_f32_16x16x32_bf16(
                false, a, false, bf, (short)0, acc[ni], false, false);
        }
    }
    if (half == 0) {
#pragma unroll
        for (int ni = 0; ni < 4; ++ni) {
            int n = n0 + ni * 16 + l15;
            float bv = out_b[n];
#pragma unroll
            for (int r = 0; r < NB; ++r)
                out[(size_t)r * VOCAB + n] = acc[ni][r] + bv;
        }
    }
}

// ---------------------------------------------------------------------------
extern "C" void kernel_launch(void* const* d_in, const int* in_sizes, int n_in,
                              void* d_out, int out_size, void* d_ws, size_t ws_size,
                              hipStream_t stream)
{
    const int*   x      = (const int*)  d_in[0];
    const float* embedW = (const float*)d_in[1];
    const float* qkv_w  = (const float*)d_in[2];
    const float* qkv_b  = (const float*)d_in[3];
    const float* out_w  = (const float*)d_in[4];
    const float* out_b  = (const float*)d_in[5];
    float* out = (float*)d_out;

    char* ws = (char*)d_ws;
    size_t off = 0;
    auto take = [&](size_t bytes) {
        char* p = ws + off;
        off = (off + bytes + 255) & ~(size_t)255;
        return p;
    };
    bf16_t* hbf    = (bf16_t*)take((size_t)NTOK * EMBED * 2);       // 32 MB
    bf16_t* qkvwt  = (bf16_t*)take((size_t)3072 * EMBED * 2);       //  6 MB
    bf16_t* owt    = (bf16_t*)take((size_t)VOCAB * EMBED * 2);      // 64 MB
    bf16_t* kbf    = (bf16_t*)take((size_t)NTOK * EMBED * 2);       // 32 MB
    bf16_t* vbf    = (bf16_t*)take((size_t)NTOK * EMBED * 2);       // 32 MB
    float*  hlast  = (float*) take((size_t)NB * EMBED * 4);
    float*  qlast  = (float*) take((size_t)NB * EMBED * 4);
    float*  logits = (float*) take((size_t)NB * SEQ * 4);
    float*  probs  = (float*) take((size_t)NB * SEQ * 4);
    float*  part   = (float*) take((size_t)KSPLIT * NB * EMBED * 4); // 256 KB
    bf16_t* alast  = (bf16_t*)take((size_t)NB * EMBED * 2);

    gather_embed_kernel<<<NTOK, 256, 0, stream>>>(x, embedW, hbf, hlast);
    transpose_bf16_kernel<<<3072, 256, 0, stream>>>(qkv_w, qkvwt, EMBED, 3072);
    transpose_bf16_kernel<<<VOCAB, 256, 0, stream>>>(out_w, owt, EMBED, VOCAB);
    gemm_kv_wmma_kernel<<<dim3(NTOK / 256, 2048 / 128), 256, 0, stream>>>(
        hbf, qkvwt + (size_t)1024 * EMBED, qkv_b + 1024, kbf, vbf);
    qlast_gemv_kernel<<<(NB * EMBED) / 256, 256, 0, stream>>>(hlast, qkvwt, qkv_b, qlast);
    logits_kernel<<<NTOK / 256, 256, 0, stream>>>(qlast, kbf, logits);
    softmax_kernel<<<NB, 256, 0, stream>>>(logits, probs);
    attnout_partial_kernel<<<dim3(NB, EMBED / 256, KSPLIT), 256, 0, stream>>>(probs, vbf, part);
    attnout_reduce_kernel<<<(NB * EMBED) / 256, 256, 0, stream>>>(part, hlast, alast);
    final_gemm_wmma_kernel<<<(VOCAB + 511) / 512, 256, 0, stream>>>(alast, owt, out_b, out);
}
